// MuxGNN_10239202033918
// MI455X (gfx1250) — matrix operand order repaired and checked by reading, not compiled, not run on hardware
//
#include <hip/hip_runtime.h>
#include <stddef.h>
#include <stdint.h>

// Problem constants (from reference)
#define NN 100000
#define RR 3
#define EE 1600000
#define DD 128   // F == D == 128
#define AA 32

typedef __attribute__((ext_vector_type(16))) __bf16 v16bf;
typedef __attribute__((ext_vector_type(8)))  float  v8f;

union BF16x16 { v16bf v; unsigned short s[16]; uint4 u[2]; };

__device__ __forceinline__ unsigned short f2bf_bits(float f) {
    return __builtin_bit_cast(unsigned short, (__bf16)f);   // native v_cvt, RNE
}

__device__ __forceinline__ void atomAddF(float* p, float v) {
    // lowers to global_atomic_add_f32 (no-return form; result unused)
    (void)__hip_atomic_fetch_add(p, v, __ATOMIC_RELAXED, __HIP_MEMORY_SCOPE_AGENT);
}

// ---------------------------------------------------------------- degrees
__global__ void deg_kernel(const int* __restrict__ src, const int* __restrict__ dst,
                           float* __restrict__ degO, float* __restrict__ degI) {
    int r = blockIdx.y;
    int e = blockIdx.x * blockDim.x + threadIdx.x;
    if (e < EE) {
        atomAddF(&degO[(size_t)r * NN + src[(size_t)r * EE + e]], 1.0f);
        atomAddF(&degI[(size_t)r * NN + dst[(size_t)r * EE + e]], 1.0f);
    }
}

// deg -> rsqrt(max(deg,1)) in place
__global__ void norm_kernel(float* __restrict__ degO, float* __restrict__ degI) {
    int i = blockIdx.x * blockDim.x + threadIdx.x;
    if (i < RR * NN) {
        degO[i] = rsqrtf(fmaxf(degO[i], 1.0f));
        degI[i] = rsqrtf(fmaxf(degI[i], 1.0f));
    }
}

// W[k][d] fp32 -> Wt[d][k] bf16 (so B-fragments are contiguous per output column)
__global__ void wprep_kernel(const float* __restrict__ W, unsigned short* __restrict__ Wt) {
    int i = blockIdx.x * blockDim.x + threadIdx.x;   // i = d*128 + k
    if (i < DD * DD) {
        int d = i >> 7, k = i & 127;
        Wt[i] = f2bf_bits(W[(size_t)k * DD + d]);
    }
}

// ------------------------------------------------------- edge scatter-add
// one wave per edge; lane l owns 4 consecutive floats; agg[r][dst] += x[r][src]*normO
__global__ void scatter_kernel(const float* __restrict__ x, size_t xstride,
                               const int* __restrict__ src, const int* __restrict__ dst,
                               const float* __restrict__ normO, float* __restrict__ agg) {
    int r    = blockIdx.y;
    int lane = threadIdx.x & 31;
    int e    = blockIdx.x * 8 + (threadIdx.x >> 5);
    if (e >= EE) return;
    int s  = src[(size_t)r * EE + e];
    int dv = dst[(size_t)r * EE + e];
    float nw = normO[(size_t)r * NN + s];
    const float4* xr = (const float4*)(x + (size_t)r * xstride + (size_t)s * DD);
    float4 v = xr[lane];
    float* ap = agg + ((size_t)r * NN + (size_t)dv) * DD + lane * 4;
    atomAddF(ap + 0, v.x * nw);
    atomAddF(ap + 1, v.y * nw);
    atomAddF(ap + 2, v.z * nw);
    atomAddF(ap + 3, v.w * nw);
}

// ------------------------------------------------------- WMMA GEMM + bias + ELU
// hout[r][n][d] = elu( (agg[r][n][:]*normI[r][n]) @ W + b )
// grid.x = N/16 row tiles, grid.y = R; block = 256 (8 waves), wave w -> column tile w
__global__ void gemm_kernel(const float* __restrict__ agg, const float* __restrict__ normI,
                            const unsigned short* __restrict__ Wt, const float* __restrict__ bias,
                            float* __restrict__ hout) {
    int r       = blockIdx.y;
    int wave    = threadIdx.x >> 5;
    int lane    = threadIdx.x & 31;
    int m       = lane & 15;          // A row within tile / D column within tile
    int khalf   = lane >> 4;
    int rowbase = blockIdx.x * 16;
    int arow_i  = rowbase + m;
    int col     = wave * 16 + m;      // N index for B and C/D

    float ninv = normI[(size_t)r * NN + arow_i];
    const float* arow = agg + ((size_t)r * NN + arow_i) * DD;
    const unsigned short* wrow = Wt + (size_t)col * DD;

    v8f c = {};
#pragma unroll
    for (int kc = 0; kc < 4; ++kc) {
        int ko = kc * 32;
        // ---- A fragment: elems e -> K = ko + khalf*16 + e
        const float4* p0 = (const float4*)(arow + ko + khalf * 16);
        const float4* p1 = (const float4*)(arow + ko + khalf * 16 + 8);
        float4 a0 = p0[0], a1 = p0[1], a2 = p1[0], a3 = p1[1];
        BF16x16 A;
        A.v[0]  = (__bf16)(a0.x * ninv); A.v[1]  = (__bf16)(a0.y * ninv);
        A.v[2]  = (__bf16)(a0.z * ninv); A.v[3]  = (__bf16)(a0.w * ninv);
        A.v[4]  = (__bf16)(a1.x * ninv); A.v[5]  = (__bf16)(a1.y * ninv);
        A.v[6]  = (__bf16)(a1.z * ninv); A.v[7]  = (__bf16)(a1.w * ninv);
        A.v[8]  = (__bf16)(a2.x * ninv); A.v[9]  = (__bf16)(a2.y * ninv);
        A.v[10] = (__bf16)(a2.z * ninv); A.v[11] = (__bf16)(a2.w * ninv);
        A.v[12] = (__bf16)(a3.x * ninv); A.v[13] = (__bf16)(a3.y * ninv);
        A.v[14] = (__bf16)(a3.z * ninv); A.v[15] = (__bf16)(a3.w * ninv);
        // ---- B fragment: elems e -> K = ko + khalf*16 + e, column = col (16 contiguous bf16)
        const uint4* q = (const uint4*)(wrow + ko + khalf * 16);
        BF16x16 B;
        B.u[0] = q[0]; B.u[1] = q[1];
        c = __builtin_amdgcn_wmma_f32_16x16x32_bf16(false, A.v, false, B.v,
                                                    (short)0, c, false, false);
    }

    float bb = bias[col];
#pragma unroll
    for (int v = 0; v < 8; ++v) {
        int mm = khalf * 8 + v;                 // C/D row: M = v + 8*(lane>>4)
        float xv = c[v] + bb;
        float o  = xv > 0.0f ? xv : (__expf(xv) - 1.0f);   // ELU(alpha=1)
        hout[((size_t)r * NN + rowbase + mm) * DD + col] = o;
    }
}

// ------------------------------------------------------- semantic attention
// one node per 96-thread block. finalLayout: 0 -> out[r][n][d], 1 -> out[n][r][d]
__global__ void sem_kernel(const float* __restrict__ hg, const float* __restrict__ s1,
                           const float* __restrict__ s2, float* __restrict__ out,
                           int finalLayout) {
    __shared__ float hl[RR][DD];
    __shared__ float th[RR][AA];
    __shared__ float sc[RR][RR];
    __shared__ float at[RR][RR];

    int n = blockIdx.x;
    int t = threadIdx.x;   // 0..95

    for (int idx = t; idx < RR * DD; idx += 96) {
        int rr = idx >> 7, dd = idx & 127;
        hl[rr][dd] = hg[((size_t)rr * NN + n) * DD + dd];
    }
    __syncthreads();

    {   // tanh(h @ s1): thread (r = t>>5, a = t&31) does a 128-length dot
        int rr = t >> 5, a = t & 31;
        float acc = 0.0f;
#pragma unroll 4
        for (int d = 0; d < DD; ++d)
            acc += hl[rr][d] * s1[((size_t)rr * DD + d) * AA + a];
        th[rr][a] = tanhf(acc);
    }
    __syncthreads();

    if (t < RR * RR) {   // scores = th @ s2
        int rr = t / RR, b = t % RR;
        float acc = 0.0f;
#pragma unroll
        for (int a = 0; a < AA; ++a)
            acc += th[rr][a] * s2[((size_t)rr * AA + a) * RR + b];
        sc[rr][b] = acc;
    }
    __syncthreads();

    if (t < RR) {        // softmax over relation axis (per output column b)
        int b = t;
        float mx = fmaxf(sc[0][b], fmaxf(sc[1][b], sc[2][b]));
        float e0 = __expf(sc[0][b] - mx);
        float e1 = __expf(sc[1][b] - mx);
        float e2 = __expf(sc[2][b] - mx);
        float inv = 1.0f / (e0 + e1 + e2);
        at[0][b] = e0 * inv; at[1][b] = e1 * inv; at[2][b] = e2 * inv;
    }
    __syncthreads();

    for (int idx = t; idx < RR * DD; idx += 96) {
        int ro = idx >> 7, dd = idx & 127;
        float v = at[ro][0] * hl[0][dd] + at[ro][1] * hl[1][dd] + at[ro][2] * hl[2][dd];
        if (finalLayout)
            out[(size_t)n * (RR * DD) + (size_t)ro * DD + dd] = v;
        else
            out[((size_t)ro * NN + n) * DD + dd] = v;
    }
}

// ---------------------------------------------------------------- launch
extern "C" void kernel_launch(void* const* d_in, const int* in_sizes, int n_in,
                              void* d_out, int out_size, void* d_ws, size_t ws_size,
                              hipStream_t stream) {
    const float* feat = (const float*)d_in[0];
    const int*   src  = (const int*)  d_in[1];
    const int*   dst  = (const int*)  d_in[2];
    const float* W0   = (const float*)d_in[3];
    const float* b0   = (const float*)d_in[4];
    const float* s1_0 = (const float*)d_in[5];
    const float* s2_0 = (const float*)d_in[6];
    const float* W1   = (const float*)d_in[7];
    const float* b1   = (const float*)d_in[8];
    const float* s1_1 = (const float*)d_in[9];
    const float* s2_1 = (const float*)d_in[10];
    float* outp = (float*)d_out;

    // workspace carve-up (256B aligned)
    char* base = (char*)d_ws;
    size_t off = 0;
    auto carve = [&](size_t bytes) -> void* {
        void* p = base + off;
        off = (off + bytes + 255) & ~(size_t)255;
        return p;
    };
    const size_t rnd = (size_t)RR * NN * DD;
    float* degO = (float*)carve((size_t)RR * NN * 4);   // -> norm_out
    float* degI = (float*)carve((size_t)RR * NN * 4);   // -> norm_in
    float* agg  = (float*)carve(rnd * 4);
    float* hg   = (float*)carve(rnd * 4);               // GCN+ELU output [R][N][D]
    float* hs   = (float*)carve(rnd * 4);               // sem output of layer 0 [R][N][D]
    unsigned short* Wt0 = (unsigned short*)carve((size_t)DD * DD * 2);
    unsigned short* Wt1 = (unsigned short*)carve((size_t)DD * DD * 2);
    (void)ws_size; (void)in_sizes; (void)n_in; (void)out_size;

    // ---- degrees + norms (shared by both layers)
    hipMemsetAsync(degO, 0, (size_t)RR * NN * 4, stream);
    hipMemsetAsync(degI, 0, (size_t)RR * NN * 4, stream);
    deg_kernel<<<dim3(EE / 256, RR), 256, 0, stream>>>(src, dst, degO, degI);
    norm_kernel<<<(RR * NN + 255) / 256, 256, 0, stream>>>(degO, degI);

    // ---- weight transpose + bf16
    wprep_kernel<<<(DD * DD + 255) / 256, 256, 0, stream>>>(W0, Wt0);
    wprep_kernel<<<(DD * DD + 255) / 256, 256, 0, stream>>>(W1, Wt1);

    // ---- layer 0
    hipMemsetAsync(agg, 0, rnd * 4, stream);
    scatter_kernel<<<dim3(EE / 8, RR), 256, 0, stream>>>(feat, (size_t)0, src, dst, degO, agg);
    gemm_kernel<<<dim3(NN / 16, RR), 256, 0, stream>>>(agg, degI, Wt0, b0, hg);
    sem_kernel<<<NN, 96, 0, stream>>>(hg, s1_0, s2_0, hs, 0);

    // ---- layer 1
    hipMemsetAsync(agg, 0, rnd * 4, stream);
    scatter_kernel<<<dim3(EE / 8, RR), 256, 0, stream>>>(hs, (size_t)NN * DD, src, dst, degO, agg);
    gemm_kernel<<<dim3(NN / 16, RR), 256, 0, stream>>>(agg, degI, Wt1, b1, hg);
    sem_kernel<<<NN, 96, 0, stream>>>(hg, s1_1, s2_1, outp, 1);
}
